// MoELayer_12919261626674
// MI455X (gfx1250) — matrix operand order repaired and checked by reading, not compiled, not run on hardware
//
#include <hip/hip_runtime.h>
#include <hip/hip_fp16.h>
#include <math.h>

#define D_MODEL 512
#define D_FF    2048
#define NEXP    8
#define TM      32   // tokens per block tile (2 WMMA m-tiles)

typedef __attribute__((ext_vector_type(16))) _Float16     v16h;
typedef __attribute__((ext_vector_type(8)))  float        v8f;
typedef __attribute__((ext_vector_type(4)))  unsigned int u32x4;
typedef __attribute__((ext_vector_type(4)))  float        f32x4;

union Frag16 { v16h v; u32x4 q[2]; };

// ---------------- utility kernels ----------------

__global__ void zero_f32_kernel(float* __restrict__ p, long n4) {
  for (long i = (long)blockIdx.x * blockDim.x + threadIdx.x; i < n4;
       i += (long)gridDim.x * blockDim.x) {
    f32x4 z = {0.f, 0.f, 0.f, 0.f};
    ((f32x4*)p)[i] = z;
  }
}

__global__ void zero_i32_kernel(int* __restrict__ p, int n) {
  int i = blockIdx.x * blockDim.x + threadIdx.x;
  if (i < n) p[i] = 0;
}

// ---------------- gating: one wave per token ----------------

__global__ void gate_kernel(const float* __restrict__ x,
                            const float* __restrict__ gw, int T,
                            int* __restrict__ counts,
                            int* __restrict__ topi, float* __restrict__ topw) {
  int wave = threadIdx.x >> 5;
  int lane = threadIdx.x & 31;
  int t = blockIdx.x * 8 + wave;
  if (t >= T) return;
  float acc[NEXP];
#pragma unroll
  for (int e = 0; e < NEXP; e++) acc[e] = 0.f;
  const float* xr = x + (long)t * D_MODEL;
  for (int j = lane; j < D_MODEL; j += 32) {
    float xv = xr[j];
#pragma unroll
    for (int e = 0; e < NEXP; e++) acc[e] += xv * gw[j * NEXP + e];
  }
#pragma unroll
  for (int e = 0; e < NEXP; e++) {
#pragma unroll
    for (int off = 16; off > 0; off >>= 1) acc[e] += __shfl_xor(acc[e], off, 32);
  }
  if (lane == 0) {
    int i0 = 0; float l0 = acc[0];
    for (int e = 1; e < NEXP; e++) if (acc[e] > l0) { l0 = acc[e]; i0 = e; }
    int i1 = -1; float l1 = -3.0e38f;
    for (int e = 0; e < NEXP; e++)
      if (e != i0 && acc[e] > l1) { l1 = acc[e]; i1 = e; }
    float w0 = 1.f / (1.f + expf(l1 - l0));  // top-2 renormalized softmax
    float w1 = 1.f - w0;
    topi[t * 2 + 0] = i0; topi[t * 2 + 1] = i1;
    topw[t * 2 + 0] = w0; topw[t * 2 + 1] = w1;
    atomicAdd(&counts[i0], 1);
    atomicAdd(&counts[i1], 1);
  }
}

__global__ void scan_kernel(const int* __restrict__ counts,
                            int* __restrict__ offsets, int* __restrict__ cursor) {
  if (threadIdx.x == 0) {
    int s = 0;
    for (int e = 0; e < NEXP; e++) { offsets[e] = s; cursor[e] = s; s += counts[e]; }
  }
}

__global__ void place_kernel(int T, const int* __restrict__ topi,
                             const float* __restrict__ topw,
                             int* __restrict__ cursor,
                             int* __restrict__ slot_token,
                             float* __restrict__ slot_w) {
  int t = blockIdx.x * blockDim.x + threadIdx.x;
  if (t >= T) return;
#pragma unroll
  for (int j = 0; j < 2; j++) {
    int e = topi[t * 2 + j];
    int s = atomicAdd(&cursor[e], 1);
    slot_token[s] = t;
    slot_w[s] = topw[t * 2 + j];
  }
}

// ------- pack weights into WMMA-B fragment-major f16 -------
// B-operand layout (32x16, 16-bit): element i of lane half h holds K = 8*h + (i&7) + (i>=8?16:0),
// N = lane&15.  packed: (((e*(K/32)+kt)*(N/16)+nt)*32 + lane)*16 + i

__global__ void pack_weights_kernel(const float* __restrict__ W,
                                    _Float16* __restrict__ Wp, int K, int N) {
  long total = (long)NEXP * K * N;
  int ntiles = N / 16, ktiles = K / 32;
  for (long idx = (long)blockIdx.x * blockDim.x + threadIdx.x; idx < total;
       idx += (long)gridDim.x * blockDim.x) {
    int  i    = (int)(idx & 15);
    long r    = idx >> 4;
    int  lane = (int)(r & 31); r >>= 5;
    int  nt   = (int)(r % ntiles); r /= ntiles;
    int  kt   = (int)(r % ktiles); r /= ktiles;
    int  e    = (int)r;
    int  half = lane >> 4, l16 = lane & 15;
    int  k = kt * 32 + 8 * half + (i & 7) + ((i >> 3) << 4);
    int  n = nt * 16 + l16;
    Wp[idx] = (_Float16)W[((long)e * K + k) * N + n];
  }
}

// ---------------- fused expert FFN ----------------

__launch_bounds__(256)
__global__ void moe_ffn_kernel(const float* __restrict__ x,
                               const _Float16* __restrict__ w1p,
                               const float* __restrict__ b1,
                               const _Float16* __restrict__ w2p,
                               const float* __restrict__ b2,
                               const int* __restrict__ counts,
                               const int* __restrict__ offsets,
                               const int* __restrict__ slot_token,
                               const float* __restrict__ slot_w,
                               float* __restrict__ out, int tilesPerExpert) {
  __shared__ alignas(16) _Float16 As[TM * D_MODEL];  // 32 KB
  __shared__ alignas(16) _Float16 Hs[TM * D_FF];     // 128 KB -> 2 WGs / 320KB WGP

  int e  = blockIdx.x / tilesPerExpert;
  int mt = blockIdx.x % tilesPerExpert;
  int ne = counts[e];
  int row0 = mt * TM;
  if (row0 >= ne) return;
  int base = offsets[e] + row0;
  int tid = threadIdx.x;

  // gather TM token rows of x -> f16 in LDS (zero-pad partial tile)
  for (int idx = tid; idx < TM * D_MODEL / 4; idx += 256) {
    int r  = idx / (D_MODEL / 4);
    int c4 = idx % (D_MODEL / 4);
    f32x4 v = {0.f, 0.f, 0.f, 0.f};
    if (row0 + r < ne) {
      int t = slot_token[base + r];
      v = *(const f32x4*)(x + (long)t * D_MODEL + c4 * 4);
    }
    _Float16* dst = &As[r * D_MODEL + c4 * 4];
    dst[0] = (_Float16)v.x; dst[1] = (_Float16)v.y;
    dst[2] = (_Float16)v.z; dst[3] = (_Float16)v.w;
  }
  __syncthreads();

  int wave = tid >> 5, lane = tid & 31;
  int half = lane >> 4, l16 = lane & 15;

  // ---- GEMM1: H[32, 2048]; wave owns 256 cols, two passes of 128 cols ----
  const _Float16* w1e = w1p + (long)e * (D_MODEL / 32) * (D_FF / 16) * 512;
#pragma unroll 1
  for (int p = 0; p < 2; ++p) {
    int ntBase = wave * 16 + p * 8;  // tile index of first owned column tile
    v8f acc[2][8];
#pragma unroll
    for (int mi = 0; mi < 2; mi++)
#pragma unroll
      for (int nt = 0; nt < 8; nt++)
#pragma unroll
        for (int j = 0; j < 8; j++) acc[mi][nt][j] = 0.f;

    Frag16 bb[2][8];
#pragma unroll
    for (int nt = 0; nt < 8; ++nt) {  // preload kt = 0
      const _Float16* bp = w1e + (((long)0 * (D_FF / 16) + ntBase + nt) * 32 + lane) * 16;
      bb[0][nt].q[0] = *(const u32x4*)(bp);
      bb[0][nt].q[1] = *(const u32x4*)(bp + 8);
    }
#pragma unroll 2
    for (int kt = 0; kt < D_MODEL / 32; ++kt) {
      int cur = kt & 1, nxt = cur ^ 1;
      if (kt + 1 < D_MODEL / 32) {  // prefetch next K-slab's B fragments
#pragma unroll
        for (int nt = 0; nt < 8; ++nt) {
          const _Float16* bp =
              w1e + (((long)(kt + 1) * (D_FF / 16) + ntBase + nt) * 32 + lane) * 16;
          bb[nxt][nt].q[0] = *(const u32x4*)(bp);
          bb[nxt][nt].q[1] = *(const u32x4*)(bp + 8);
        }
      }
      Frag16 a0, a1;  // A 16x32: half element i -> K = 8*half + i + (i>=8?8:0)
      const _Float16* ap0 = &As[l16 * D_MODEL + kt * 32 + 8 * half];
      const _Float16* ap1 = ap0 + 16 * D_MODEL;
      a0.q[0] = *(const u32x4*)(ap0);
      a0.q[1] = *(const u32x4*)(ap0 + 16);
      a1.q[0] = *(const u32x4*)(ap1);
      a1.q[1] = *(const u32x4*)(ap1 + 16);
#pragma unroll
      for (int nt = 0; nt < 8; ++nt) {  // B fragment reused for both m-tiles
        acc[0][nt] = __builtin_amdgcn_wmma_f32_16x16x32_f16(
            false, a0.v, false, bb[cur][nt].v, (short)0, acc[0][nt], false, false);
        acc[1][nt] = __builtin_amdgcn_wmma_f32_16x16x32_f16(
            false, a1.v, false, bb[cur][nt].v, (short)0, acc[1][nt], false, false);
      }
    }
    // exact GELU epilogue -> Hs (f16)
#pragma unroll
    for (int nt = 0; nt < 8; ++nt) {
      int col = (ntBase + nt) * 16 + l16;
      float bbias = b1[e * D_FF + col];
#pragma unroll
      for (int mi = 0; mi < 2; ++mi)
#pragma unroll
        for (int v = 0; v < 8; ++v) {
          int m = mi * 16 + v + 8 * half;
          float hv = acc[mi][nt][v] + bbias;
          hv = 0.5f * hv * (1.f + erff(hv * 0.70710678118654752f));
          Hs[m * D_FF + col] = (_Float16)hv;
        }
    }
  }
  __syncthreads();

  // ---- GEMM2: Y[32, 512]; wave owns 64 cols ----
  v8f acc2[2][4];
#pragma unroll
  for (int mi = 0; mi < 2; mi++)
#pragma unroll
    for (int nt = 0; nt < 4; nt++)
#pragma unroll
      for (int j = 0; j < 8; j++) acc2[mi][nt][j] = 0.f;

  const _Float16* w2e = w2p + (long)e * (D_FF / 32) * (D_MODEL / 16) * 512;
  Frag16 cb[2][4];
#pragma unroll
  for (int nt = 0; nt < 4; ++nt) {  // preload kt = 0
    const _Float16* bp =
        w2e + (((long)0 * (D_MODEL / 16) + wave * 4 + nt) * 32 + lane) * 16;
    cb[0][nt].q[0] = *(const u32x4*)(bp);
    cb[0][nt].q[1] = *(const u32x4*)(bp + 8);
  }
#pragma unroll 2
  for (int kt = 0; kt < D_FF / 32; ++kt) {
    int cur = kt & 1, nxt = cur ^ 1;
    if (kt + 1 < D_FF / 32) {
#pragma unroll
      for (int nt = 0; nt < 4; ++nt) {
        const _Float16* bp =
            w2e + (((long)(kt + 1) * (D_MODEL / 16) + wave * 4 + nt) * 32 + lane) * 16;
        cb[nxt][nt].q[0] = *(const u32x4*)(bp);
        cb[nxt][nt].q[1] = *(const u32x4*)(bp + 8);
      }
    }
    Frag16 a0, a1;
    const _Float16* ap0 = &Hs[l16 * D_FF + kt * 32 + 8 * half];
    const _Float16* ap1 = ap0 + 16 * D_FF;
    a0.q[0] = *(const u32x4*)(ap0);
    a0.q[1] = *(const u32x4*)(ap0 + 16);
    a1.q[0] = *(const u32x4*)(ap1);
    a1.q[1] = *(const u32x4*)(ap1 + 16);
#pragma unroll
    for (int nt = 0; nt < 4; ++nt) {
      acc2[0][nt] = __builtin_amdgcn_wmma_f32_16x16x32_f16(
          false, a0.v, false, cb[cur][nt].v, (short)0, acc2[0][nt], false, false);
      acc2[1][nt] = __builtin_amdgcn_wmma_f32_16x16x32_f16(
          false, a1.v, false, cb[cur][nt].v, (short)0, acc2[1][nt], false, false);
    }
  }

  // scatter: out[t] += w * (y + b2); exactly 2 commutative adds per element
#pragma unroll
  for (int nt = 0; nt < 4; ++nt) {
    int col = wave * 64 + nt * 16 + l16;
    float bbias = b2[e * D_MODEL + col];
#pragma unroll
    for (int mi = 0; mi < 2; ++mi)
#pragma unroll
      for (int v = 0; v < 8; ++v) {
        int m = mi * 16 + v + 8 * half;
        if (row0 + m < ne) {
          int slot = base + m;
          int t = slot_token[slot];
          float w = slot_w[slot];
          atomicAdd(&out[(long)t * D_MODEL + col], w * (acc2[mi][nt][v] + bbias));
        }
      }
  }
}

// ---------------- host side ----------------

extern "C" void kernel_launch(void* const* d_in, const int* in_sizes, int n_in,
                              void* d_out, int out_size, void* d_ws, size_t ws_size,
                              hipStream_t stream) {
  const float* x  = (const float*)d_in[0];
  const float* gw = (const float*)d_in[1];
  const float* w1 = (const float*)d_in[2];
  const float* b1 = (const float*)d_in[3];
  const float* w2 = (const float*)d_in[4];
  const float* b2 = (const float*)d_in[5];
  float* out = (float*)d_out;

  int T = in_sizes[0] / D_MODEL;  // 32768

  char* ws = (char*)d_ws;
  size_t off = 0;
  auto alloc = [&](size_t bytes) -> void* {
    off = (off + 255) & ~(size_t)255;
    void* p = ws + off;
    off += bytes;
    return p;
  };
  int*      counts     = (int*)alloc(NEXP * sizeof(int));
  int*      offsets    = (int*)alloc(NEXP * sizeof(int));
  int*      cursor     = (int*)alloc(NEXP * sizeof(int));
  int*      topi       = (int*)alloc((size_t)T * 2 * sizeof(int));
  float*    topw       = (float*)alloc((size_t)T * 2 * sizeof(float));
  int*      slot_token = (int*)alloc((size_t)T * 2 * sizeof(int));
  float*    slot_w     = (float*)alloc((size_t)T * 2 * sizeof(float));
  _Float16* w1p        = (_Float16*)alloc((size_t)NEXP * D_MODEL * D_FF * sizeof(_Float16));
  _Float16* w2p        = (_Float16*)alloc((size_t)NEXP * D_FF * D_MODEL * sizeof(_Float16));

  // init
  zero_f32_kernel<<<4096, 256, 0, stream>>>(out, (long)out_size / 4);
  zero_i32_kernel<<<1, 32, 0, stream>>>(counts, NEXP);

  // routing
  gate_kernel<<<(T + 7) / 8, 256, 0, stream>>>(x, gw, T, counts, topi, topw);
  scan_kernel<<<1, 32, 0, stream>>>(counts, offsets, cursor);
  place_kernel<<<(T + 255) / 256, 256, 0, stream>>>(T, topi, topw, cursor,
                                                    slot_token, slot_w);

  // weight packing (f16, fragment-major)
  pack_weights_kernel<<<4096, 256, 0, stream>>>(w1, w1p, D_MODEL, D_FF);
  pack_weights_kernel<<<4096, 256, 0, stream>>>(w2, w2p, D_FF, D_MODEL);

  // fused expert FFN
  int tilesPerExpert = (T + TM - 1) / TM;  // 1024 worst case; blocks early-exit
  moe_ffn_kernel<<<NEXP * tilesPerExpert, 256, 0, stream>>>(
      x, w1p, b1, w2p, b2, counts, offsets, slot_token, slot_w, out,
      tilesPerExpert);
}
